// Net_13692355740240
// MI455X (gfx1250) — compile-verified, matrix operand order checked by reading, not run on hardware
//
#include <hip/hip_runtime.h>
#include <hip/hip_bf16.h>

#define VOCAB 50257
#define EMB   256
#define HID   512
#define G4    2048      // 4*HID
#define BTOK  32768
#define TILE_M 16

#define LDA_E (EMB + 8)   // padded LDS row stride (ushorts) for e tile
#define LDA_H (HID + 8)   // padded LDS row stride (ushorts) for h tile

typedef __attribute__((ext_vector_type(16))) __bf16 v16bf;
typedef __attribute__((ext_vector_type(8)))  float  v8f;
typedef __attribute__((ext_vector_type(4)))  unsigned int u32x4;
typedef __attribute__((ext_vector_type(8)))  int    i32x8;
typedef __attribute__((ext_vector_type(4)))  int    i32x4;

union FragB16 { v16bf v; unsigned int u[8]; };

#if __has_builtin(__builtin_amdgcn_tensor_load_to_lds) && \
    __has_builtin(__builtin_amdgcn_s_wait_tensorcnt)
#define USE_TDM 1
#else
#define USE_TDM 0
#endif

__device__ __forceinline__ unsigned short f32_to_bf16(float f) {
    unsigned int u = __float_as_uint(f);
    unsigned int r = u + 0x7FFFu + ((u >> 16) & 1u);   // round-to-nearest-even
    return (unsigned short)(r >> 16);
}

__device__ __forceinline__ float fast_sigmoid(float x) {
    return 1.0f / (1.0f + __expf(-x));
}
__device__ __forceinline__ float fast_tanh(float x) {
    return 1.0f - 2.0f / (__expf(2.0f * x) + 1.0f);
}

// K index of the low element of the bf16 pair held in fragment dword v,
// for lane-half `half` (CDNA5 ISA, 16-bit A-matrix 16x32 layout).
__device__ __forceinline__ int pair_k(int v, int half) {
    return (v < 4 ? 2 * v : 16 + 2 * (v - 4)) + half * 8;
}

__device__ __forceinline__ unsigned int rfl(unsigned int v) {
    return (unsigned int)__builtin_amdgcn_readfirstlane((int)v);
}

// ---------------------------------------------------------------------------
// Prep: fp32 -> bf16 weight conversion (done once per launch, stays in L2),
// and fold the two gate biases.
// ---------------------------------------------------------------------------
__global__ __launch_bounds__(256) void prep_weights(
    const float* __restrict__ w_ih,
    const float* __restrict__ w_dense,
    const float* __restrict__ b_ih,
    const float* __restrict__ b_hh,
    unsigned short* __restrict__ w_ih_bf,
    unsigned short* __restrict__ w_dense_bf,
    float* __restrict__ bias_g)
{
    int i = blockIdx.x * blockDim.x + threadIdx.x;
    if (i < G4 * EMB)  w_ih_bf[i]    = f32_to_bf16(w_ih[i]);
    if (i < EMB * HID) w_dense_bf[i] = f32_to_bf16(w_dense[i]);
    if (i < G4)        bias_g[i]     = b_ih[i] + b_hh[i];
}

// ---------------------------------------------------------------------------
// Fused: TDM-gathered embed tile -> LSTM step (h0=c0=0 => i,g,o gates only)
// -> dense head. One block per 16 tokens, 8 waves.
// ---------------------------------------------------------------------------
__global__ __launch_bounds__(256) void lstm_fused(
    const int* __restrict__ x,
    const float* __restrict__ embed_w,
    const unsigned short* __restrict__ w_ih_bf,     // [2048][256] bf16
    const unsigned short* __restrict__ w_dense_bf,  // [256][512]  bf16
    const float* __restrict__ bias_g,               // [2048] = b_ih + b_hh
    const float* __restrict__ b_dense,              // [256]
    float* __restrict__ out)                        // [32768][256] f32
{
    __shared__ float          eStage[TILE_M * EMB]; // f32 staging for TDM gather
    __shared__ unsigned short eA[TILE_M * LDA_E];   // e tile, bf16 row-major
    __shared__ unsigned short hB[TILE_M * LDA_H];   // h tile, bf16 row-major
    __shared__ int toks[TILE_M];

    const int tid   = threadIdx.x;
    const int blk   = blockIdx.x;
    const int wave  = tid >> 5;
    const int lane  = tid & 31;
    const int half  = lane >> 4;
    const int nloc  = lane & 15;

    if (tid < TILE_M) toks[tid] = x[blk * TILE_M + tid];
    __syncthreads();

    // ---- Stage 0: gather 16 embedding rows into LDS ------------------------
#if USE_TDM
    if (wave == 0) {
        // Tensor DMA descriptor, gather mode (CDNA5 ISA):
        // 16 rows selected by 16-bit indices, each row 256 f32 = 1 KB,
        // packed contiguously at eStage.
        unsigned long long ga = (unsigned long long)(uintptr_t)embed_w;
        unsigned int ga_lo   = rfl((unsigned int)ga);
        unsigned int ga_hi   = rfl((unsigned int)(ga >> 32));
        unsigned int lds_off = rfl((unsigned int)(uintptr_t)(void*)&eStage[0]);

        u32x4 g0;
        g0.x = 1u | (1u << 31);                 // count=1, gather_mode=1, idx=16-bit
        g0.y = lds_off;                         // lds_addr
        g0.z = ga_lo;                           // global_addr[31:0]
        g0.w = (ga_hi & 0x01FFFFFFu) | (2u << 30); // global_addr[56:32], type=2

        i32x8 g1;
        g1[0] = (int)(2u << 16);                // wg_mask=0, data_size=2 (4 bytes)
        g1[1] = (int)((EMB & 0xFFFFu) << 16);   // tensor_dim0[15:0] = 256
        g1[2] = (int)((VOCAB & 0xFFFFu) << 16); // tensor_dim1[15:0]
        g1[3] = (int)((VOCAB >> 16) & 0xFFFFu)  // tensor_dim1[31:16]
              | (int)((EMB & 0xFFFFu) << 16);   // tile_dim0 = 256
        g1[4] = TILE_M;                         // tile_dim1 = 16 valid indices
        g1[5] = EMB;                            // tensor_dim0_stride = 256
        g1[6] = 0;
        g1[7] = 0;

        int idx_pack[8];
#pragma unroll
        for (int j = 0; j < 8; ++j) {
            unsigned int lo = (unsigned int)toks[2 * j]     & 0xFFFFu;
            unsigned int hi = (unsigned int)toks[2 * j + 1] & 0xFFFFu;
            idx_pack[j] = (int)rfl(lo | (hi << 16));
        }
        i32x4 g2 = { idx_pack[0], idx_pack[1], idx_pack[2], idx_pack[3] };
        i32x4 g3 = { idx_pack[4], idx_pack[5], idx_pack[6], idx_pack[7] };

#if defined(__clang_major__) && (__clang_major__ >= 23)
        i32x8 g4 = { 0, 0, 0, 0, 0, 0, 0, 0 }; // amdgpu-toolchain 6-arg form
        __builtin_amdgcn_tensor_load_to_lds(g0, g1, g2, g3, g4, 0);
#else
        __builtin_amdgcn_tensor_load_to_lds(g0, g1, g2, g3, 0); // ROCm 7.2 5-arg
#endif
        __builtin_amdgcn_s_wait_tensorcnt(0);
    }
    __syncthreads();   // TDM data visible in LDS to all waves

    // convert staged f32 tile -> bf16 A tile (padded stride)
    for (int i = tid; i < TILE_M * EMB; i += 256) {
        int r = i >> 8;            // / EMB
        int c = i & (EMB - 1);
        eA[r * LDA_E + c] = f32_to_bf16(eStage[i]);
    }
#else
    for (int i = tid; i < TILE_M * EMB; i += 256) {
        int r = i >> 8;
        int c = i & (EMB - 1);
        float v = embed_w[(long long)toks[r] * EMB + c];
        eA[r * LDA_E + c] = f32_to_bf16(v);
    }
#endif
    __syncthreads();

    // ---- Stage 1: i,g,o gate tiles + activations -> h tile (bf16, LDS) ----
    for (int t = 0; t < 4; ++t) {
        const int htile   = wave * 4 + t;   // 0..31
        const int colbase = htile * 16;
        v8f acc_i = {}, acc_g = {}, acc_o = {};

        for (int kb = 0; kb < EMB / 32; ++kb) {
            FragB16 a, bi, bg, bo;
            const int rowi = 0 * HID + colbase + nloc;   // i-gate row of w_ih
            const int rowg = 2 * HID + colbase + nloc;   // g-gate row
            const int rowo = 3 * HID + colbase + nloc;   // o-gate row
#pragma unroll
            for (int v = 0; v < 8; ++v) {
                int k = kb * 32 + pair_k(v, half);
                a.u[v]  = *(const unsigned int*)&eA[nloc * LDA_E + k];
                bi.u[v] = *(const unsigned int*)&w_ih_bf[rowi * EMB + k];
                bg.u[v] = *(const unsigned int*)&w_ih_bf[rowg * EMB + k];
                bo.u[v] = *(const unsigned int*)&w_ih_bf[rowo * EMB + k];
            }
            acc_i = __builtin_amdgcn_wmma_f32_16x16x32_bf16(
                        false, a.v, false, bi.v, (short)0, acc_i, false, false);
            acc_g = __builtin_amdgcn_wmma_f32_16x16x32_bf16(
                        false, a.v, false, bg.v, (short)0, acc_g, false, false);
            acc_o = __builtin_amdgcn_wmma_f32_16x16x32_bf16(
                        false, a.v, false, bo.v, (short)0, acc_o, false, false);
        }

        const int col = colbase + nloc;
        const float bias_i  = bias_g[0 * HID + col];
        const float bias_gg = bias_g[2 * HID + col];
        const float bias_o  = bias_g[3 * HID + col];
#pragma unroll
        for (int r = 0; r < 8; ++r) {
            int row  = r + half * 8;     // C/D layout: lanes 16-31 hold M=r+8
            float iv = acc_i[r] + bias_i;
            float gv = acc_g[r] + bias_gg;
            float ov = acc_o[r] + bias_o;
            float cc = fast_sigmoid(iv) * fast_tanh(gv);   // c0 = 0
            float hh = fast_sigmoid(ov) * fast_tanh(cc);
            hB[row * LDA_H + col] = f32_to_bf16(hh);
        }
    }
    __syncthreads();

    // ---- Stage 2: out = h @ w_dense^T + b_dense ---------------------------
    for (int t = 0; t < 2; ++t) {
        const int otile   = wave * 2 + t;   // 0..15
        const int colbase = otile * 16;
        v8f acc = {};

        for (int kb = 0; kb < HID / 32; ++kb) {
            FragB16 a, b;
#pragma unroll
            for (int v = 0; v < 8; ++v) {
                int k = kb * 32 + pair_k(v, half);
                a.u[v] = *(const unsigned int*)&hB[nloc * LDA_H + k];
                b.u[v] = *(const unsigned int*)&w_dense_bf[(colbase + nloc) * HID + k];
            }
            acc = __builtin_amdgcn_wmma_f32_16x16x32_bf16(
                      false, a.v, false, b.v, (short)0, acc, false, false);
        }

        const int col  = colbase + nloc;
        const float bb = b_dense[col];
#pragma unroll
        for (int r = 0; r < 8; ++r) {
            int row = r + half * 8;
            // streamed output: non-temporal, don't pollute L2
            __builtin_nontemporal_store(
                acc[r] + bb,
                &out[((long long)blk * TILE_M + row) * EMB + col]);
        }
    }
}

// ---------------------------------------------------------------------------
extern "C" void kernel_launch(void* const* d_in, const int* in_sizes, int n_in,
                              void* d_out, int out_size, void* d_ws, size_t ws_size,
                              hipStream_t stream) {
    const int*   x        = (const int*)  d_in[0];
    const float* embed_w  = (const float*)d_in[1];
    const float* w_ih     = (const float*)d_in[2];
    // d_in[3] = w_hh : unused, h0 == 0
    const float* b_ih     = (const float*)d_in[4];
    const float* b_hh     = (const float*)d_in[5];
    const float* w_dense  = (const float*)d_in[6];
    const float* b_dense  = (const float*)d_in[7];
    float*       out      = (float*)d_out;

    // Workspace layout: [w_ih bf16 | w_dense bf16 | bias_g f32]  (~1.3 MB)
    unsigned short* w_ih_bf    = (unsigned short*)d_ws;
    unsigned short* w_dense_bf = w_ih_bf + (size_t)G4 * EMB;
    float*          bias_g     = (float*)(w_dense_bf + (size_t)EMB * HID);

    prep_weights<<<(G4 * EMB + 255) / 256, 256, 0, stream>>>(
        w_ih, w_dense, b_ih, b_hh, w_ih_bf, w_dense_bf, bias_g);

    lstm_fused<<<BTOK / TILE_M, 256, 0, stream>>>(
        x, embed_w, w_ih_bf, w_dense_bf, bias_g, b_dense, out);
}
